// MultiQueryAttention_54606214201756
// MI455X (gfx1250) — matrix operand order repaired and checked by reading, not compiled, not run on hardware
//
#include <hip/hip_runtime.h>
#include <hip/hip_bf16.h>

// ---------------- problem constants ----------------
#define DIMC  2048
#define NH    16
#define NKV   2
#define DH    128
#define GQ    (NH / NKV)      // 8
#define TSEQ  2048
#define NB    2
#define MROWS (NB * TSEQ)     // 4096
#define ATT_SCALE 0.08838834764831845f          // 1/sqrt(128)
#define LOG2E     1.4426950408889634f
#define QSCALE    (ATT_SCALE * LOG2E)           // fold into Q projection
#define CLIP2     144.26950408889634f           // 100 * log2(e): clip in exp2 domain

typedef __attribute__((ext_vector_type(16))) _Float16 v16h;
typedef __attribute__((ext_vector_type(8)))  _Float16 v8h;
typedef __attribute__((ext_vector_type(8)))  float    v8f;

#if __has_builtin(__builtin_amdgcn_exp2f)
#define EXP2F(x) __builtin_amdgcn_exp2f(x)
#else
#define EXP2F(x) __expf((x) * 0.6931471805599453f)
#endif

// ---- CDNA5 async global->LDS staging (guarded so the build cannot break) ----
#if __has_builtin(__builtin_amdgcn_global_load_async_to_lds_b128)
#define ATTN_ASYNC 1
typedef int v4i_gcc __attribute__((vector_size(4 * sizeof(int))));
typedef __attribute__((address_space(1))) v4i_gcc gv4i;  // global (AS1)
typedef __attribute__((address_space(3))) v4i_gcc lv4i;  // LDS (AS3)
#define ASYNC_B128(g, l)                                                       \
  __builtin_amdgcn_global_load_async_to_lds_b128((gv4i*)(g), (lv4i*)(l), 0, 0)
#if __has_builtin(__builtin_amdgcn_s_wait_asynccnt)
#define WAIT_ASYNC(n) __builtin_amdgcn_s_wait_asynccnt(n)
#else
#define WAIT_ASYNC(n) asm volatile("s_wait_asynccnt %0" ::"i"(n) : "memory")
#endif
#else
#define ATTN_ASYNC 0
#endif

static __device__ __forceinline__ v16h join16(v8h lo, v8h hi) {
  v16h r;
#pragma unroll
  for (int i = 0; i < 8; ++i) { r[i] = lo[i]; r[i + 8] = hi[i]; }
  return r;
}

static __device__ __forceinline__ v8f wmma_fp16(v16h a, v16h b, v8f c) {
  // D = A(16x32 f16) * B(32x16 f16) + C(16x16 f32)
  return __builtin_amdgcn_wmma_f32_16x16x32_f16(false, a, false, b, (short)0, c,
                                                false, false);
}

// DPP-based reductions across the 16 lanes of a DPP row (wave32 = 2 rows),
// matching the WMMA C-layout (row m lives in one 16-lane half).
#define DPPF(x, ctrl)                                                          \
  __int_as_float(__builtin_amdgcn_update_dpp(0, __float_as_int(x), (ctrl),     \
                                             0xF, 0xF, true))

static __device__ __forceinline__ float redmax16(float x) {
  x = fmaxf(x, DPPF(x, 0xB1));   // quad_perm xor1
  x = fmaxf(x, DPPF(x, 0x4E));   // quad_perm xor2
  x = fmaxf(x, DPPF(x, 0x141));  // row_half_mirror
  x = fmaxf(x, DPPF(x, 0x140));  // row_mirror
  return x;
}
static __device__ __forceinline__ float redsum16(float x) {
  x += DPPF(x, 0xB1);
  x += DPPF(x, 0x4E);
  x += DPPF(x, 0x141);
  x += DPPF(x, 0x140);
  return x;
}

// ---------------- prep kernels ----------------
__global__ void cast_f32_to_f16_kernel(const float* __restrict__ in,
                                       _Float16* __restrict__ out, int n) {
  int i = blockIdx.x * blockDim.x + threadIdx.x;
  if (i < n) out[i] = (_Float16)in[i];
}

// W: [K,N] f32 row-major  ->  WT: [N,K] f16 row-major (coalesced writes)
__global__ void transpose_to_f16_kernel(const float* __restrict__ W,
                                        _Float16* __restrict__ WT,
                                        int K, int N) {
  int idx = blockIdx.x * blockDim.x + threadIdx.x;
  if (idx < K * N) {
    int n = idx / K;
    int k = idx - n * K;
    WT[(size_t)n * K + k] = (_Float16)W[(size_t)k * N + n];
  }
}

// ---------------- WMMA GEMM: C = A[M,K] * BT[N,K]^T + bias ----------------
// wave tile 64x64 (16 accumulators), block = 8 waves (4x2) -> 256x128 tile
// MODE 0: Q  -> f16, *QSCALE (scale + exp2-domain), layout [b, h, t, d]
// MODE 1: K  -> f16, layout [b, hk, t, d]
// MODE 2: V  -> f16, layout [b, hk, d, t]   (transposed for PV B-fragments)
// MODE 3: f32 row-major [M, N] (final output)
template <int MODE>
__global__ __launch_bounds__(256) void gemm_wmma_kernel(
    const _Float16* __restrict__ A, const _Float16* __restrict__ BT,
    const float* __restrict__ bias, void* __restrict__ Cout,
    int M, int N, int K) {
  const int lane = threadIdx.x & 31;
  const int wave = threadIdx.x >> 5;
  const int hf   = lane >> 4;    // lane half (0/1)
  const int lm   = lane & 15;
  const int wr   = wave >> 1;    // 0..3
  const int wc   = wave & 1;     // 0..1
  const int m0   = blockIdx.y * 256 + wr * 64;
  const int n0   = blockIdx.x * 128 + wc * 64;

  v8f acc[4][4];
#pragma unroll
  for (int mt = 0; mt < 4; ++mt)
#pragma unroll
    for (int nt = 0; nt < 4; ++nt) acc[mt][nt] = {};

  for (int k0 = 0; k0 < K; k0 += 32) {
    v16h af[4], bf[4];
#pragma unroll
    for (int mt = 0; mt < 4; ++mt) {
      const _Float16* ap = A + (size_t)(m0 + mt * 16 + lm) * K + k0 + hf * 16;
      af[mt] = join16(*(const v8h*)ap, *(const v8h*)(ap + 8));
    }
#pragma unroll
    for (int nt = 0; nt < 4; ++nt) {
      const _Float16* bp = BT + (size_t)(n0 + nt * 16 + lm) * K + k0 + hf * 16;
      bf[nt] = *(const v16h*)bp;
    }
#pragma unroll
    for (int mt = 0; mt < 4; ++mt)
#pragma unroll
      for (int nt = 0; nt < 4; ++nt)
        acc[mt][nt] = wmma_fp16(af[mt], bf[nt], acc[mt][nt]);
  }

#pragma unroll
  for (int mt = 0; mt < 4; ++mt)
#pragma unroll
    for (int nt = 0; nt < 4; ++nt)
#pragma unroll
      for (int r = 0; r < 8; ++r) {
        const int row = m0 + mt * 16 + hf * 8 + r;   // C layout: M = r + 8*half
        const int col = n0 + nt * 16 + lm;           // C layout: N = lane&15
        float v = acc[mt][nt][r] + bias[col];
        if constexpr (MODE == 0) {
          int bi = row / TSEQ, t = row - bi * TSEQ;
          int h = col / DH, d = col - h * DH;
          ((_Float16*)Cout)[(((size_t)(bi * NH + h)) * TSEQ + t) * DH + d] =
              (_Float16)(v * QSCALE);
        } else if constexpr (MODE == 1) {
          int bi = row / TSEQ, t = row - bi * TSEQ;
          int hk = col / DH, d = col - hk * DH;
          ((_Float16*)Cout)[(((size_t)(bi * NKV + hk)) * TSEQ + t) * DH + d] =
              (_Float16)v;
        } else if constexpr (MODE == 2) {
          int bi = row / TSEQ, t = row - bi * TSEQ;
          int hk = col / DH, d = col - hk * DH;
          ((_Float16*)Cout)[(((size_t)(bi * NKV + hk)) * DH + d) * TSEQ + t] =
              (_Float16)v;
        } else {
          ((float*)Cout)[(size_t)row * N + col] = v;
        }
      }
}

// ---------------- flash attention (online softmax, all-WMMA) ----------------
// grid = (TSEQ/128, NH, NB), block = 256 (8 waves, 16 queries per wave).
// K/V tiles (64 keys) are staged in LDS once per block iteration via async
// global->LDS DMA (double-buffered); all 8 waves share them, cutting global
// K/V traffic 8x. 32 WMMAs per 64-key iteration.
__global__ __launch_bounds__(256) void attn_kernel(
    const _Float16* __restrict__ Q, const _Float16* __restrict__ Km,
    const _Float16* __restrict__ VT, _Float16* __restrict__ Out) {
  __shared__ alignas(16) _Float16 Klds[2][64][128];  // 32KB: keys x d
  __shared__ alignas(16) _Float16 Vlds[2][128][64];  // 32KB: d x keys (V^T)
  __shared__ alignas(16) _Float16 Plds[8][16][64];   // 16KB: P staging per wave

  const int tid  = threadIdx.x;
  const int lane = tid & 31;
  const int wave = tid >> 5;
  const int hf   = lane >> 4;
  const int lm   = lane & 15;
  const int h    = blockIdx.y;
  const int bi   = blockIdx.z;
  const int hk   = h / GQ;
  const int q0   = blockIdx.x * 128 + wave * 16;

  const _Float16* Qp = Q  + ((size_t)(bi * NH + h))   * TSEQ * DH;
  const _Float16* Kp = Km + ((size_t)(bi * NKV + hk)) * TSEQ * DH;
  const _Float16* Vp = VT + ((size_t)(bi * NKV + hk)) * DH * TSEQ;

#if ATTN_ASYNC
  // issue one 64-key K/V tile into LDS buffer `bufi` (8 asyncs per wave)
  auto issue_tile = [&](int kt, int bufi) {
#pragma unroll
    for (int c = 0; c < 4; ++c) {
      const int ck   = tid + c * 256;          // 0..1023 16B chunks
      const int krow = ck >> 4, koff = (ck & 15) << 3;
      ASYNC_B128(Kp + (size_t)(kt + krow) * DH + koff, &Klds[bufi][krow][koff]);
      const int vrow = ck >> 3, voff = (ck & 7) << 3;
      ASYNC_B128(Vp + (size_t)vrow * TSEQ + kt + voff, &Vlds[bufi][vrow][voff]);
    }
  };
  issue_tile(0, 0);
#endif

  // Q fragments for 16 rows x 128 (4 k-chunks of 32), hoisted out of the loop
  v16h aq[4];
#pragma unroll
  for (int kc = 0; kc < 4; ++kc) {
    const _Float16* p = Qp + (size_t)(q0 + lm) * DH + kc * 32 + hf * 16;
    aq[kc] = join16(*(const v8h*)p, *(const v8h*)(p + 8));
  }

  v8f o[8];
#pragma unroll
  for (int dt = 0; dt < 8; ++dt) o[dt] = {};
  float mrow[8], lrow[8];
#pragma unroll
  for (int r = 0; r < 8; ++r) { mrow[r] = -1e30f; lrow[r] = 0.0f; }

  int bufi = 0;
  for (int kt = 0; kt < TSEQ; kt += 64) {
#if ATTN_ASYNC
    if (kt + 64 < TSEQ) {
      issue_tile(kt + 64, bufi ^ 1);  // prefetch next tile into other buffer
      WAIT_ASYNC(8);                  // previous tile done; 8 new in flight
    } else {
      WAIT_ASYNC(0);
    }
    __syncthreads();                  // tile `kt` visible to all waves
#else
    // synchronous fallback staging (single buffer)
#pragma unroll
    for (int c = 0; c < 4; ++c) {
      const int ck   = tid + c * 256;
      const int krow = ck >> 4, koff = (ck & 15) << 3;
      *(v8h*)&Klds[0][krow][koff] =
          *(const v8h*)(Kp + (size_t)(kt + krow) * DH + koff);
      const int vrow = ck >> 3, voff = (ck & 7) << 3;
      *(v8h*)&Vlds[0][vrow][voff] =
          *(const v8h*)(Vp + (size_t)vrow * TSEQ + kt + voff);
    }
    __syncthreads();
#endif

    // ---- S2 = (Q*scale*log2e) * K^T for 64 keys (four 16x16 tiles) ----
    v8f s[4];
#pragma unroll
    for (int tt = 0; tt < 4; ++tt) {
      s[tt] = {};
      const _Float16* kb = &Klds[bufi][tt * 16 + lm][hf * 16];
#pragma unroll
      for (int kc = 0; kc < 4; ++kc) {
        v16h bk = *(const v16h*)(kb + kc * 32);
        s[tt] = wmma_fp16(aq[kc], bk, s[tt]);
      }
    }
    // clip (exp2 domain: +-100*log2e), matches reference clip(scores,+-100)
#pragma unroll
    for (int tt = 0; tt < 4; ++tt)
#pragma unroll
      for (int r = 0; r < 8; ++r)
        s[tt][r] = fminf(fmaxf(s[tt][r], -CLIP2), CLIP2);

    // ---- chunk row-max (DPP tree over 16 lanes of the half) ----
    float mc[8];
#pragma unroll
    for (int r = 0; r < 8; ++r) {
      float a0 = fmaxf(s[0][r], s[1][r]);
      float a1 = fmaxf(s[2][r], s[3][r]);
      mc[r] = redmax16(fmaxf(a0, a1));
    }

    float alpha[8];
#pragma unroll
    for (int r = 0; r < 8; ++r) {
      float mn = fmaxf(mrow[r], mc[r]);
      alpha[r] = EXP2F(mrow[r] - mn);
      mrow[r]  = mn;
    }

    // ---- P = exp2(S2 - m), per-lane partial sums then one DPP reduction ----
    float psum[8];
#pragma unroll
    for (int r = 0; r < 8; ++r) psum[r] = 0.0f;
#pragma unroll
    for (int tt = 0; tt < 4; ++tt)
#pragma unroll
      for (int r = 0; r < 8; ++r) {
        float p = EXP2F(s[tt][r] - mrow[r]);
        s[tt][r] = p;
        psum[r] += p;
      }
#pragma unroll
    for (int r = 0; r < 8; ++r) psum[r] = redsum16(psum[r]);
#pragma unroll
    for (int r = 0; r < 8; ++r) lrow[r] = lrow[r] * alpha[r] + psum[r];

    // rescale O accumulators
#pragma unroll
    for (int dt = 0; dt < 8; ++dt)
#pragma unroll
      for (int r = 0; r < 8; ++r) o[dt][r] *= alpha[r];

    // ---- stage P (C-layout) -> LDS -> A-fragments (same wave: LDS in-order) ----
#pragma unroll
    for (int tt = 0; tt < 4; ++tt)
#pragma unroll
      for (int r = 0; r < 8; ++r)
        Plds[wave][hf * 8 + r][tt * 16 + lm] = (_Float16)s[tt][r];

    v16h pa[2];
#pragma unroll
    for (int kc2 = 0; kc2 < 2; ++kc2) {
      const _Float16* pp = &Plds[wave][lm][kc2 * 32 + hf * 16];
      pa[kc2] = join16(*(const v8h*)pp, *(const v8h*)(pp + 8));
    }

    // ---- O += P * V  (B-frags from the shared V^T LDS tile) ----
#pragma unroll
    for (int dt = 0; dt < 8; ++dt) {
      const _Float16* vb = &Vlds[bufi][dt * 16 + lm][hf * 16];
#pragma unroll
      for (int kc2 = 0; kc2 < 2; ++kc2) {
        v16h bv = *(const v16h*)(vb + kc2 * 32);
        o[dt] = wmma_fp16(pa[kc2], bv, o[dt]);
      }
    }

    __syncthreads();  // all waves done reading this buffer before reuse
#if ATTN_ASYNC
    bufi ^= 1;
#endif
  }

  // ---- normalize and write [b*T, NH*DH] f16 ----
  float inv[8];
#pragma unroll
  for (int r = 0; r < 8; ++r) inv[r] = 1.0f / lrow[r];
#pragma unroll
  for (int dt = 0; dt < 8; ++dt)
#pragma unroll
    for (int r = 0; r < 8; ++r) {
      const int row = q0 + hf * 8 + r;
      const int col = h * DH + dt * 16 + lm;
      Out[(size_t)(bi * TSEQ + row) * (NH * DH) + col] =
          (_Float16)(o[dt][r] * inv[r]);
    }
}

// ---------------- launch ----------------
extern "C" void kernel_launch(void* const* d_in, const int* in_sizes, int n_in,
                              void* d_out, int out_size, void* d_ws, size_t ws_size,
                              hipStream_t stream) {
  const float* x  = (const float*)d_in[0];
  const float* Wq = (const float*)d_in[1];
  const float* bq = (const float*)d_in[2];
  const float* Wk = (const float*)d_in[3];
  const float* bk = (const float*)d_in[4];
  const float* Wv = (const float*)d_in[5];
  const float* bv = (const float*)d_in[6];
  const float* Wo = (const float*)d_in[7];
  const float* bo = (const float*)d_in[8];
  float* out = (float*)d_out;

  char* ws = (char*)d_ws;
  const size_t MB = 1024ull * 1024ull;
  _Float16* xh  = (_Float16*)(ws + 0 * MB);    // 16 MiB  x as f16 [4096,2048]
  _Float16* WqT = (_Float16*)(ws + 16 * MB);   //  8 MiB  [2048,2048]
  _Float16* WkT = (_Float16*)(ws + 24 * MB);   //  1 MiB  [256,2048]
  _Float16* WvT = (_Float16*)(ws + 25 * MB);   //  1 MiB  [256,2048]
  _Float16* WoT = (_Float16*)(ws + 26 * MB);   //  8 MiB  [2048,2048]
  _Float16* Qh  = (_Float16*)(ws + 34 * MB);   // 16 MiB  [b,h,t,d]
  _Float16* Kh  = (_Float16*)(ws + 50 * MB);   //  2 MiB  [b,hk,t,d]
  _Float16* VTh = (_Float16*)(ws + 52 * MB);   //  2 MiB  [b,hk,d,t]
  _Float16* AO  = (_Float16*)(ws + 54 * MB);   // 16 MiB  attn out [4096,2048]
  // total 70 MiB of workspace

  { // x -> f16
    int n = MROWS * DIMC;
    cast_f32_to_f16_kernel<<<n / 256, 256, 0, stream>>>(x, xh, n);
  }
  { // weight transposes -> f16 [N,K]
    int n = DIMC * DIMC;
    transpose_to_f16_kernel<<<(n + 255) / 256, 256, 0, stream>>>(Wq, WqT, DIMC, DIMC);
    transpose_to_f16_kernel<<<(n + 255) / 256, 256, 0, stream>>>(Wo, WoT, DIMC, DIMC);
    int nk = DIMC * (NKV * DH);
    transpose_to_f16_kernel<<<(nk + 255) / 256, 256, 0, stream>>>(Wk, WkT, DIMC, NKV * DH);
    transpose_to_f16_kernel<<<(nk + 255) / 256, 256, 0, stream>>>(Wv, WvT, DIMC, NKV * DH);
  }

  // projections (all WMMA): block tile 256x128
  gemm_wmma_kernel<0><<<dim3(DIMC / 128, MROWS / 256), 256, 0, stream>>>(
      xh, WqT, bq, Qh, MROWS, DIMC, DIMC);
  gemm_wmma_kernel<1><<<dim3((NKV * DH) / 128, MROWS / 256), 256, 0, stream>>>(
      xh, WkT, bk, Kh, MROWS, NKV * DH, DIMC);
  gemm_wmma_kernel<2><<<dim3((NKV * DH) / 128, MROWS / 256), 256, 0, stream>>>(
      xh, WvT, bv, VTh, MROWS, NKV * DH, DIMC);

  // flash attention (async LDS-staged K/V)
  attn_kernel<<<dim3(TSEQ / 128, NH, NB), 256, 0, stream>>>(Qh, Kh, VTh, AO);

  // output projection -> f32
  gemm_wmma_kernel<3><<<dim3(DIMC / 128, MROWS / 256), 256, 0, stream>>>(
      AO, WoT, bo, out, MROWS, DIMC, DIMC);
}
